// KNNMultiHeadAttention_36258113912835
// MI455X (gfx1250) — matrix operand order repaired and checked by reading, not compiled, not run on hardware
//
#include <hip/hip_runtime.h>
#include <hip/hip_bf16.h>

#define NUM_HEADS 16
#define D_MODEL   1024
#define D_HEAD    64
#define TOP_K     64
#define BATCH     2
#define SEQ       2048
#define TOKENS    (BATCH * SEQ)      // 4096
#define NQKV      (3 * D_MODEL)      // 3072

typedef __attribute__((ext_vector_type(16))) __bf16 bf16x16;
typedef __attribute__((ext_vector_type(8)))  float  f32x8;
typedef __attribute__((ext_vector_type(4)))  unsigned int u32x4;

// GCC-vector int4 + address-space-qualified pointers matching the async
// global->LDS builtin's expected parameter types (from hipcc diagnostics).
typedef int v4i_gcc __attribute__((vector_size(16)));
typedef __attribute__((address_space(1))) v4i_gcc* as1_v4i;
typedef __attribute__((address_space(3))) v4i_gcc* as3_v4i;

// Detect CDNA5 async global->LDS builtins (toolchain-dependent).
#if defined(__has_builtin)
#  if __has_builtin(__builtin_amdgcn_global_load_async_to_lds_b128)
#    define HAVE_ASYNC_LDS 1
#  endif
#endif
#ifndef HAVE_ASYNC_LDS
#  define HAVE_ASYNC_LDS 0
#endif

__device__ __forceinline__ void wait_async0() {
#if defined(__has_builtin) && __has_builtin(__builtin_amdgcn_s_wait_asynccnt)
  __builtin_amdgcn_s_wait_asynccnt(0);
#else
  asm volatile("s_wait_asynccnt 0x0" ::: "memory");
#endif
}

// ---------------------------------------------------------------------------
// WMMA wrapper: D = A(16x32 bf16) * B(32x16 bf16) + C(16x16 f32)
// ---------------------------------------------------------------------------
__device__ __forceinline__ f32x8 wmma_bf16(bf16x16 a, bf16x16 b, f32x8 c) {
  return __builtin_amdgcn_wmma_f32_16x16x32_bf16(
      /*neg_a=*/false, a, /*neg_b=*/false, b,
      /*c_mod=*/(short)0, c, /*reuse_a=*/false, /*reuse_b=*/false);
}

// ---------------------------------------------------------------------------
// Fragment loaders (wave32; contiguous k order, same for A and B)
// A 16x32 bf16: lane&15 = row M; lane<16 holds K in {0..15},
// lane>=16 holds K in {16..31}; VGPR pair j -> K = 2j+half*16
// ---------------------------------------------------------------------------
__device__ __forceinline__ bf16x16
load_a_frag(const __bf16* A, int lda, int row0, int k0, int lane) {
  int half = lane >> 4;
  int m    = lane & 15;
  const __bf16* p = A + (long)(row0 + m) * lda + k0;
  bf16x16 a;
#pragma unroll
  for (int j = 0; j < 8; ++j) {
    int kk = 2 * j + half * 16;
    a[2 * j]     = p[kk];
    a[2 * j + 1] = p[kk + 1];
  }
  return a;
}

// B 32x16 bf16 where B[kk][n] = W[n][kk] (i.e. B = W^T, W row-major NxK):
// lane&15 = col n; VGPR j holds K = 2j,2j+1 (+16 for lanes 16..31)
__device__ __forceinline__ bf16x16
load_bT_frag(const __bf16* W, int ldw, int n0, int k0, int lane) {
  int half = lane >> 4;
  int n    = lane & 15;
  const __bf16* p = W + (long)(n0 + n) * ldw + k0 + half * 16;
  bf16x16 b;
#pragma unroll
  for (int j = 0; j < 8; ++j) {
    b[2 * j]     = p[2 * j];
    b[2 * j + 1] = p[2 * j + 1];
  }
  return b;
}

// B 32x16 bf16 from LDS-staged row-major tile vst[32][D_HEAD]
__device__ __forceinline__ bf16x16
load_b_lds(const __bf16* vst, int n0, int lane) {
  int half = lane >> 4;
  int n    = (lane & 15) + n0;
  bf16x16 b;
#pragma unroll
  for (int j = 0; j < 8; ++j) {
    int kk = 2 * j + half * 16;
    b[2 * j]     = vst[kk * D_HEAD + n];
    b[2 * j + 1] = vst[(kk + 1) * D_HEAD + n];
  }
  return b;
}

// ---------------------------------------------------------------------------
// Kernel 1: fp32 -> bf16 conversion
// ---------------------------------------------------------------------------
__global__ void cvt_f32_bf16(const float* __restrict__ in,
                             __bf16* __restrict__ out, long n) {
  long i = (long)blockIdx.x * blockDim.x + threadIdx.x;
  long stride = (long)gridDim.x * blockDim.x;
  for (; i < n; i += stride) out[i] = (__bf16)in[i];
}

// ---------------------------------------------------------------------------
// Kernel 2: QKV GEMM: qkv[4096,3072] = x_bf[4096,1024] @ Wqkv_bf[3072,1024]^T
//            + bias; Q columns pre-scaled by 1/sqrt(D_HEAD).
// Block = 8 waves (32x8). Wave tile 32x64 (2x4 WMMA tiles); block tile 128x128.
// ---------------------------------------------------------------------------
__global__ __launch_bounds__(256) void
qkv_gemm(const __bf16* __restrict__ X, const __bf16* __restrict__ W,
         const float* __restrict__ bias, __bf16* __restrict__ out) {
  int lane = threadIdx.x;
  int wave = threadIdx.y;
  int half = lane >> 4;
  int wm = wave & 3, wn = wave >> 2;
  int row0 = blockIdx.y * 128 + wm * 32;
  int col0 = blockIdx.x * 128 + wn * 64;

  f32x8 acc[2][4];
#pragma unroll
  for (int mi = 0; mi < 2; ++mi)
#pragma unroll
    for (int n = 0; n < 4; ++n) acc[mi][n] = 0.0f;

  for (int k0 = 0; k0 < D_MODEL; k0 += 32) {
    bf16x16 a0 = load_a_frag(X, D_MODEL, row0,      k0, lane);
    bf16x16 a1 = load_a_frag(X, D_MODEL, row0 + 16, k0, lane);
#pragma unroll
    for (int n = 0; n < 4; ++n) {
      bf16x16 b = load_bT_frag(W, D_MODEL, col0 + 16 * n, k0, lane);
      acc[0][n] = wmma_bf16(a0, b, acc[0][n]);
      acc[1][n] = wmma_bf16(a1, b, acc[1][n]);
    }
  }

#pragma unroll
  for (int mi = 0; mi < 2; ++mi)
#pragma unroll
    for (int n = 0; n < 4; ++n)
#pragma unroll
      for (int v = 0; v < 8; ++v) {
        int r = row0 + mi * 16 + half * 8 + v;
        int c = col0 + n * 16 + (lane & 15);
        float val = acc[mi][n][v] + bias[c];
        if ((c % 192) < D_HEAD) val *= 0.125f;  // fold 1/sqrt(64) into Q
        out[(long)r * NQKV + c] = (__bf16)val;
      }
}

// ---------------------------------------------------------------------------
// Kernel 3: fused KNN attention for one (b, h, 16-query block).
// Phase 1: scores(16x2048) = Q @ K^T via WMMA -> LDS (fp16).
// Phase 2: per-query top-64 threshold (binary search) + softmax stats.
// Phase 3: out(16x64) = masked-softmax(P) @ V via WMMA; V tiles staged into
//          LDS with coalesced b128 copies (async global->LDS when available),
//          cross-wave reduction through LDS.
// ---------------------------------------------------------------------------
__global__ __launch_bounds__(256) void
knn_attn(const __bf16* __restrict__ qkv, __bf16* __restrict__ attn) {
  __shared__ _Float16 sc[16][SEQ];               // 64 KB score block
  __shared__ float outbuf[16][D_HEAD];           //  4 KB cross-wave accum
  __shared__ float row_m[16], row_thr[16], row_inv[16];
  __shared__ __align__(16) __bf16 vstage[8][32 * D_HEAD];  // 32 KB V staging

  int lane = threadIdx.x;
  int wave = threadIdx.y;
  int half = lane >> 4;
  int tid  = wave * 32 + lane;

  int qb = blockIdx.x;          // query block (0..127)
  int h  = blockIdx.y;          // head
  int b  = blockIdx.z;          // batch
  int q0 = qb * 16;

  const __bf16* Q = qkv + (long)b * SEQ * NQKV + h * 192;
  const __bf16* K = Q + D_HEAD;
  const __bf16* V = Q + 2 * D_HEAD;

  // ---- Phase 1: scores = Q @ K^T (Q already scaled by 1/8) ----
  bf16x16 aq0 = load_a_frag(Q, NQKV, q0, 0, lane);
  bf16x16 aq1 = load_a_frag(Q, NQKV, q0, 32, lane);
  for (int t = wave; t < SEQ / 16; t += 8) {
    bf16x16 b0 = load_bT_frag(K, NQKV, t * 16, 0, lane);
    bf16x16 b1 = load_bT_frag(K, NQKV, t * 16, 32, lane);
    f32x8 c = 0.0f;
    c = wmma_bf16(aq0, b0, c);
    c = wmma_bf16(aq1, b1, c);
#pragma unroll
    for (int v = 0; v < 8; ++v)
      sc[v + half * 8][t * 16 + (lane & 15)] = (_Float16)c[v];
  }
  __syncthreads();

  // ---- Phase 2: top-64 threshold + softmax stats (2 queries per wave) ----
  for (int qi = wave; qi < 16; qi += 8) {
    float mx = -3.0e38f, mn = 3.0e38f;
    for (int i = lane; i < SEQ; i += 32) {
      float s = (float)sc[qi][i];
      mx = fmaxf(mx, s);
      mn = fminf(mn, s);
    }
#pragma unroll
    for (int o = 16; o >= 1; o >>= 1) {
      mx = fmaxf(mx, __shfl_xor(mx, o, 32));
      mn = fminf(mn, __shfl_xor(mn, o, 32));
    }
    float lo = mn, hi = mx;
    for (int it = 0; it < 24; ++it) {      // binary search for 64th largest
      float mid = 0.5f * (lo + hi);
      int cnt = 0;
      for (int i = lane; i < SEQ; i += 32)
        cnt += ((float)sc[qi][i] >= mid) ? 1 : 0;
#pragma unroll
      for (int o = 16; o >= 1; o >>= 1) cnt += __shfl_xor(cnt, o, 32);
      if (cnt >= TOP_K) lo = mid; else hi = mid;
    }
    float thr = lo;
    float ssum = 0.0f;
    for (int i = lane; i < SEQ; i += 32) {
      float s = (float)sc[qi][i];
      if (s >= thr) ssum += __expf(s - mx);
    }
#pragma unroll
    for (int o = 16; o >= 1; o >>= 1) ssum += __shfl_xor(ssum, o, 32);
    if (lane == 0) {
      row_m[qi] = mx;
      row_thr[qi] = thr;
      row_inv[qi] = 1.0f / ssum;
    }
  }
  for (int i = tid; i < 16 * D_HEAD; i += 256) ((float*)outbuf)[i] = 0.0f;
  __syncthreads();

  // ---- Phase 3: out = P @ V, keys split across waves (256 each) ----
  int m = lane & 15;
  float qm = row_m[m], qt = row_thr[m];
  f32x8 acc[4];
#pragma unroll
  for (int n = 0; n < 4; ++n) acc[n] = 0.0f;

  __bf16* vst = vstage[wave];

  for (int ks = 0; ks < 8; ++ks) {
    int kb = wave * 256 + ks * 32;
    const __bf16* vsrc = V + (long)kb * NQKV;

    // stage V[kb..kb+31][0..63] into this wave's LDS region (coalesced b128)
#if HAVE_ASYNC_LDS
#pragma unroll
    for (int t = 0; t < 8; ++t) {
      int idx  = t * 32 + lane;       // 0..255 transfers of 16B
      int row  = idx >> 3;            // 0..31
      int col8 = (idx & 7) * 8;       // 0,8,...,56
      __builtin_amdgcn_global_load_async_to_lds_b128(
          (as1_v4i)(vsrc + (long)row * NQKV + col8),
          (as3_v4i)(vst + row * D_HEAD + col8),
          0, 0);
    }
    wait_async0();  // same-wave produce->consume: ASYNCcnt==0 suffices
#else
#pragma unroll
    for (int t = 0; t < 8; ++t) {
      int idx  = t * 32 + lane;
      int row  = idx >> 3;
      int col8 = (idx & 7) * 8;
      *(u32x4*)(vst + row * D_HEAD + col8) =
          *(const u32x4*)(vsrc + (long)row * NQKV + col8);
    }
#endif

    // A fragment: masked softmax probabilities regenerated from LDS scores
    bf16x16 ap;
#pragma unroll
    for (int j = 0; j < 8; ++j) {
      int kk = kb + 2 * j + half * 16;
      float s0 = (float)sc[m][kk];
      float s1 = (float)sc[m][kk + 1];
      ap[2 * j]     = (__bf16)((s0 >= qt) ? __expf(s0 - qm) : 0.0f);
      ap[2 * j + 1] = (__bf16)((s1 >= qt) ? __expf(s1 - qm) : 0.0f);
    }
#pragma unroll
    for (int n = 0; n < 4; ++n) {
      bf16x16 bv = load_b_lds(vst, n * 16, lane);
      acc[n] = wmma_bf16(ap, bv, acc[n]);
    }
  }
#pragma unroll
  for (int n = 0; n < 4; ++n)
#pragma unroll
    for (int v = 0; v < 8; ++v)
      atomicAdd(&outbuf[v + half * 8][n * 16 + (lane & 15)], acc[n][v]);
  __syncthreads();

  // normalize + write attn[token][h*64+d]
  for (int i = tid; i < 16 * D_HEAD; i += 256) {
    int qi = i >> 6, d = i & 63;
    float val = outbuf[qi][d] * row_inv[qi];
    attn[(long)(b * SEQ + q0 + qi) * D_MODEL + h * D_HEAD + d] = (__bf16)val;
  }
}

// ---------------------------------------------------------------------------
// Kernel 4: output projection + residual:
// out[4096,1024] = x + attn_bf @ Wout_bf^T + bias   (fp32 result)
// ---------------------------------------------------------------------------
__global__ __launch_bounds__(256) void
proj_gemm(const __bf16* __restrict__ A, const __bf16* __restrict__ W,
          const float* __restrict__ bias, const float* __restrict__ xres,
          float* __restrict__ out) {
  int lane = threadIdx.x;
  int wave = threadIdx.y;
  int half = lane >> 4;
  int wm = wave & 3, wn = wave >> 2;
  int row0 = blockIdx.y * 128 + wm * 32;
  int col0 = blockIdx.x * 128 + wn * 64;

  f32x8 acc[2][4];
#pragma unroll
  for (int mi = 0; mi < 2; ++mi)
#pragma unroll
    for (int n = 0; n < 4; ++n) acc[mi][n] = 0.0f;

  for (int k0 = 0; k0 < D_MODEL; k0 += 32) {
    bf16x16 a0 = load_a_frag(A, D_MODEL, row0,      k0, lane);
    bf16x16 a1 = load_a_frag(A, D_MODEL, row0 + 16, k0, lane);
#pragma unroll
    for (int n = 0; n < 4; ++n) {
      bf16x16 b = load_bT_frag(W, D_MODEL, col0 + 16 * n, k0, lane);
      acc[0][n] = wmma_bf16(a0, b, acc[0][n]);
      acc[1][n] = wmma_bf16(a1, b, acc[1][n]);
    }
  }

#pragma unroll
  for (int mi = 0; mi < 2; ++mi)
#pragma unroll
    for (int n = 0; n < 4; ++n)
#pragma unroll
      for (int v = 0; v < 8; ++v) {
        int r = row0 + mi * 16 + half * 8 + v;
        int c = col0 + n * 16 + (lane & 15);
        out[(long)r * D_MODEL + c] =
            acc[mi][n][v] + bias[c] + xres[(long)r * D_MODEL + c];
      }
}

// ---------------------------------------------------------------------------
// Host launch
// ---------------------------------------------------------------------------
extern "C" void kernel_launch(void* const* d_in, const int* in_sizes, int n_in,
                              void* d_out, int out_size, void* d_ws,
                              size_t ws_size, hipStream_t stream) {
  const float* x       = (const float*)d_in[0];
  const float* w_qkv_w = (const float*)d_in[1];
  const float* w_qkv_b = (const float*)d_in[2];
  const float* w_out_w = (const float*)d_in[3];
  const float* w_out_b = (const float*)d_in[4];
  float* out = (float*)d_out;

  // workspace layout (bf16 buffers)
  char* ws = (char*)d_ws;
  __bf16* x_bf    = (__bf16*)(ws);                       //  8 MB
  __bf16* wqkv_bf = (__bf16*)(ws + (8u << 20));          //  6 MB
  __bf16* wout_bf = (__bf16*)(ws + (14u << 20));         //  2 MB
  __bf16* qkv_bf  = (__bf16*)(ws + (16u << 20));         // 24 MB
  __bf16* attn_bf = (__bf16*)(ws + (40u << 20));         //  8 MB

  // 1) fp32 -> bf16 conversions
  cvt_f32_bf16<<<1024, 256, 0, stream>>>(x, x_bf, (long)TOKENS * D_MODEL);
  cvt_f32_bf16<<<1024, 256, 0, stream>>>(w_qkv_w, wqkv_bf,
                                         (long)NQKV * D_MODEL);
  cvt_f32_bf16<<<1024, 256, 0, stream>>>(w_out_w, wout_bf,
                                         (long)D_MODEL * D_MODEL);

  // 2) QKV projection (4096 x 3072 x 1024)
  {
    dim3 grid(NQKV / 128, TOKENS / 128);   // 24 x 32
    dim3 block(32, 8);
    qkv_gemm<<<grid, block, 0, stream>>>(x_bf, wqkv_bf, w_qkv_b, qkv_bf);
  }

  // 3) fused KNN attention
  {
    dim3 grid(SEQ / 16, NUM_HEADS, BATCH); // 128 x 16 x 2
    dim3 block(32, 8);
    knn_attn<<<grid, block, 0, stream>>>(qkv_bf, attn_bf);
  }

  // 4) output projection + residual (4096 x 1024 x 1024)
  {
    dim3 grid(D_MODEL / 128, TOKENS / 128); // 8 x 32
    dim3 block(32, 8);
    proj_gemm<<<grid, block, 0, stream>>>(attn_bf, wout_bf, w_out_b, x, out);
  }
}
